// TGN_59201829208561
// MI455X (gfx1250) — compile-verified, matrix operand order checked by reading, not run on hardware
//
#include <hip/hip_runtime.h>
#include <hip/hip_bf16.h>

typedef __bf16 bf16;
typedef __attribute__((ext_vector_type(16))) __bf16 bf16x16;
typedef __attribute__((ext_vector_type(8)))  __bf16 bf16x8;
typedef __attribute__((ext_vector_type(8)))  float  f32x8;

// ---- problem dims ----
#define NS     16384          // nodes in subgraph
#define ES     131072         // edges
#define BS     4096           // events
#define MEMD   256
#define TDIM   128
#define MSGD   172
#define GRUIN  812
#define GRUINP 832            // padded to 26*32
#define EDGED  300
#define EDGEDP 320            // padded to 10*32
#define EMBD   256
#define G3     768            // 3*MEM gates

// ================= utility =================
__device__ inline float wave_sum(float s) {
  #pragma unroll
  for (int off = 16; off; off >>= 1) s += __shfl_xor(s, off, 32);
  return s;
}
__device__ inline float sigmoidf_(float x) { return 1.0f / (1.0f + expf(-x)); }
__device__ inline void atomicMaxF(float* addr, float v) {
  if (v >= 0.0f) atomicMax((int*)addr, __float_as_int(v));
  else atomicMin((unsigned int*)addr, __float_as_uint(v));
}

// ================= kernels =================

// f32 -> bf16 with column padding (weights)
__global__ void k_convert_pad(const float* __restrict__ src, bf16* __restrict__ dst,
                              int R, int C, int Cp) {
  size_t i = (size_t)blockIdx.x * blockDim.x + threadIdx.x;
  size_t total = (size_t)R * Cp;
  if (i >= total) return;
  int r = (int)(i / Cp), c = (int)(i % Cp);
  float v = (c < C) ? src[(size_t)r * C + c] : 0.0f;
  dst[i] = (bf16)v;
}

__global__ void k_init(float* denom, unsigned int* maxv) {
  int i = blockIdx.x * blockDim.x + threadIdx.x;
  if (i < NS * 2) denom[i] = 0.0f;
  if (i < 2) maxv[i] = 0xFF800000u;   // -inf bits
}

// gather node state, time-encode, build GRU input row (bf16, padded)
__global__ void k_build_aggr(const int* __restrict__ node_ids,
                             const int* __restrict__ mem_last_update,
                             const int* __restrict__ mem_rel_t,
                             const int* __restrict__ mem_dst_id,
                             const float* __restrict__ mem_table,
                             const float* __restrict__ mem_msg_table,
                             const float* __restrict__ mem_dir,
                             const float* __restrict__ te_w,
                             const float* __restrict__ te_b,
                             bf16* __restrict__ aggr,
                             float* __restrict__ hf,
                             bf16* __restrict__ hb,
                             float* __restrict__ luf) {
  int i = blockIdx.x;
  int nid = node_ids[i];
  int dst = mem_dst_id[nid];
  float d0 = mem_dir[2 * (size_t)nid], d1 = mem_dir[2 * (size_t)nid + 1];
  float relt = (float)mem_rel_t[nid];
  if (threadIdx.x == 0) luf[i] = (float)mem_last_update[nid];
  const float* sm = mem_table + (size_t)nid * MEMD;
  const float* dm = mem_table + (size_t)dst * MEMD;
  bf16* arow = aggr + (size_t)i * GRUINP;
  for (int c = threadIdx.x; c < GRUINP; c += blockDim.x) {
    float v;
    if (c < 256) {
      float s = sm[c], d = dm[c];
      v = s * d0 + d * d1;
      hf[(size_t)i * 256 + c] = s;
      hb[(size_t)i * 256 + c] = (bf16)s;
    } else if (c < 512) {
      int cc = c - 256; v = sm[cc] * d1 + dm[cc] * d0;
    } else if (c < 512 + MSGD) {
      v = mem_msg_table[(size_t)nid * MSGD + (c - 512)];
    } else if (c < GRUIN) {
      int td = c - (512 + MSGD);
      v = cosf(relt * te_w[td] + te_b[td]);
    } else {
      v = 0.0f;
    }
    arow[c] = (bf16)v;
  }
}

// generic WMMA GEMM: C[M,N] = A[M,K](bf16,row) * B[N,K](bf16,row)^T + bias, opt relu
// block = 128 threads (4 waves); wave -> 32 rows x 32 cols (2 M-tiles x 2 N-tiles)
// -> 8 b128 loads per 4 WMMAs, 4 independent accumulation chains
__global__ void k_gemm_bf16(const bf16* __restrict__ A, const bf16* __restrict__ B,
                            const float* __restrict__ bias, float* __restrict__ C,
                            int M, int N, int K, int relu) {
  int lane = threadIdx.x & 31;
  int wave = threadIdx.x >> 5;
  int row0 = (blockIdx.y * 4 + wave) << 5;   // 32 rows per wave
  int col0 = blockIdx.x << 5;                // 32 cols per wave
  if (row0 >= M) return;
  int l16 = lane & 15, hi = lane >> 4;
  const bf16* arow0 = A + (size_t)(row0 + l16) * K;
  const bf16* arow1 = A + (size_t)(row0 + 16 + l16) * K;
  const bf16* brow0 = B + (size_t)(col0 + l16) * K;
  const bf16* brow1 = B + (size_t)(col0 + 16 + l16) * K;
  f32x8 acc00 = {}, acc01 = {}, acc10 = {}, acc11 = {};
  int koff0 = hi * 8, koff1 = 16 + hi * 8;
  for (int kk = 0; kk < K; kk += 32) {
    bf16x16 a0, a1, b0, b1;
    ((bf16x8*)&a0)[0] = *(const bf16x8*)(arow0 + kk + koff0);
    ((bf16x8*)&a0)[1] = *(const bf16x8*)(arow0 + kk + koff1);
    ((bf16x8*)&a1)[0] = *(const bf16x8*)(arow1 + kk + koff0);
    ((bf16x8*)&a1)[1] = *(const bf16x8*)(arow1 + kk + koff1);
    ((bf16x8*)&b0)[0] = *(const bf16x8*)(brow0 + kk + koff0);
    ((bf16x8*)&b0)[1] = *(const bf16x8*)(brow0 + kk + koff1);
    ((bf16x8*)&b1)[0] = *(const bf16x8*)(brow1 + kk + koff0);
    ((bf16x8*)&b1)[1] = *(const bf16x8*)(brow1 + kk + koff1);
    acc00 = __builtin_amdgcn_wmma_f32_16x16x32_bf16(false, a0, false, b0, (short)0, acc00, false, false);
    acc01 = __builtin_amdgcn_wmma_f32_16x16x32_bf16(false, a0, false, b1, (short)0, acc01, false, false);
    acc10 = __builtin_amdgcn_wmma_f32_16x16x32_bf16(false, a1, false, b0, (short)0, acc10, false, false);
    acc11 = __builtin_amdgcn_wmma_f32_16x16x32_bf16(false, a1, false, b1, (short)0, acc11, false, false);
  }
  float bv0 = bias ? bias[col0 + l16] : 0.0f;
  float bv1 = bias ? bias[col0 + 16 + l16] : 0.0f;
  #pragma unroll
  for (int r = 0; r < 8; ++r) {
    size_t roff0 = (size_t)(row0 + r + hi * 8) * N;
    size_t roff1 = (size_t)(row0 + 16 + r + hi * 8) * N;
    float v00 = acc00[r] + bv0, v01 = acc01[r] + bv1;
    float v10 = acc10[r] + bv0, v11 = acc11[r] + bv1;
    if (relu) {
      v00 = fmaxf(v00, 0.0f); v01 = fmaxf(v01, 0.0f);
      v10 = fmaxf(v10, 0.0f); v11 = fmaxf(v11, 0.0f);
    }
    C[roff0 + col0 + l16]      = v00;
    C[roff0 + col0 + 16 + l16] = v01;
    C[roff1 + col0 + l16]      = v10;
    C[roff1 + col0 + 16 + l16] = v11;
  }
}

// GRU pointwise gates
__global__ void k_gru(const float* __restrict__ gi, const float* __restrict__ gh,
                      const float* __restrict__ hf, float* __restrict__ memf,
                      bf16* __restrict__ memb) {
  size_t i = (size_t)blockIdx.x * blockDim.x + threadIdx.x;
  if (i >= (size_t)NS * MEMD) return;
  int n = (int)(i >> 8), c = (int)(i & 255);
  const float* gir = gi + (size_t)n * G3;
  const float* ghr = gh + (size_t)n * G3;
  float r  = sigmoidf_(gir[c]       + ghr[c]);
  float z  = sigmoidf_(gir[256 + c] + ghr[256 + c]);
  float nn = tanhf(gir[512 + c] + r * ghr[512 + c]);
  float h  = hf[i];
  float m  = (1.0f - z) * nn + z * h;
  memf[i] = m;
  memb[i] = (bf16)m;
}

// per-edge feature row: [edge_msg | cos(rel*w+b) | pad] -> bf16
__global__ void k_build_ea(const int* __restrict__ edge_idx,
                           const float* __restrict__ luf,
                           const float* __restrict__ edge_t,
                           const float* __restrict__ edge_msg,
                           const float* __restrict__ te_w,
                           const float* __restrict__ te_b,
                           bf16* __restrict__ ea) {
  int e = blockIdx.x;
  int src = edge_idx[e];
  float rel = luf[src] - edge_t[e];
  bf16* row = ea + (size_t)e * EDGEDP;
  const float* msg = edge_msg + (size_t)e * MSGD;
  for (int c = threadIdx.x; c < EDGEDP; c += blockDim.x) {
    float v;
    if (c < MSGD)       v = msg[c];
    else if (c < EDGED) { int td = c - MSGD; v = cosf(rel * te_w[td] + te_b[td]); }
    else                v = 0.0f;
    row[c] = (bf16)v;
  }
}

// attention logits: one wave per (edge, head)
__global__ void k_alpha(const int* __restrict__ edge_idx,
                        const float* __restrict__ q, const float* __restrict__ kn,
                        const float* __restrict__ eproj,
                        float* __restrict__ alpha, float* __restrict__ maxv) {
  int lane = threadIdx.x & 31;
  int wv = threadIdx.x >> 5;
  int eh = blockIdx.x * 8 + wv;
  int e = eh >> 1, h = eh & 1;
  int src = edge_idx[e], dst = edge_idx[ES + e];
  const float4* qp = (const float4*)(q     + (size_t)dst * EMBD + h * 128) + lane;
  const float4* kp = (const float4*)(kn    + (size_t)src * EMBD + h * 128) + lane;
  const float4* ep = (const float4*)(eproj + (size_t)e  * EMBD + h * 128) + lane;
  float4 qv = *qp, kv = *kp, ev = *ep;
  float s = qv.x * (kv.x + ev.x) + qv.y * (kv.y + ev.y) +
            qv.z * (kv.z + ev.z) + qv.w * (kv.w + ev.w);
  s = wave_sum(s) * 0.08838834764831845f;   // 1/sqrt(128)
  if (lane == 0) {
    alpha[(size_t)e * 2 + h] = s;
    atomicMaxF(&maxv[h], s);
  }
}

// exp + segment-sum denominator
__global__ void k_expdenom(const int* __restrict__ edge_idx,
                           float* __restrict__ alpha, const float* __restrict__ maxv,
                           float* __restrict__ denom) {
  int i = blockIdx.x * blockDim.x + threadIdx.x;
  if (i >= ES * 2) return;
  int e = i >> 1, h = i & 1;
  int dst = edge_idx[ES + e];
  float ex = expf(alpha[i] - maxv[h]);
  alpha[i] = ex;
  atomicAdd(&denom[(size_t)dst * 2 + h], ex);
}

// weighted scatter-add of values into z (z pre-filled with skip projection)
__global__ void k_scatter(const int* __restrict__ edge_idx,
                          const float* __restrict__ alpha, const float* __restrict__ denom,
                          const float* __restrict__ vn, const float* __restrict__ eproj,
                          float* __restrict__ z) {
  int e = blockIdx.x, c = threadIdx.x, h = c >> 7;
  int src = edge_idx[e], dst = edge_idx[ES + e];
  float coeff = alpha[(size_t)e * 2 + h] / (denom[(size_t)dst * 2 + h] + 1e-16f);
  float val = (vn[(size_t)src * EMBD + c] + eproj[(size_t)e * EMBD + c]) * coeff;
  atomicAdd(&z[(size_t)dst * EMBD + c], val);
}

// build link-predictor input rows: cat(z[a], z[b]) as bf16  (rows 0..B-1 pos, B..2B-1 neg)
__global__ void k_cat(const int* __restrict__ node_idx, const float* __restrict__ z,
                      bf16* __restrict__ cat) {
  int r = blockIdx.x;
  int b = r & (BS - 1);
  int i0 = node_idx[b];
  int i1 = (r < BS) ? node_idx[BS + b] : node_idx[2 * BS + b];
  bf16* row = cat + (size_t)r * (2 * EMBD);
  for (int c = threadIdx.x; c < 2 * EMBD; c += blockDim.x) {
    float v = (c < EMBD) ? z[(size_t)i0 * EMBD + c] : z[(size_t)i1 * EMBD + (c - EMBD)];
    row[c] = (bf16)v;
  }
}

// final layer: dot(hidden_row, lp_fw) + lp_fb, mask, sigmoid
__global__ void k_final(const float* __restrict__ hidden, const float* __restrict__ lp_fw,
                        const float* __restrict__ lp_fb, const int* __restrict__ node_idx,
                        float* __restrict__ out) {
  int lane = threadIdx.x & 31;
  int wv = threadIdx.x >> 5;
  int r = blockIdx.x * 8 + wv;
  const float* hr = hidden + (size_t)r * EMBD;
  float s = 0.0f;
  #pragma unroll
  for (int c = lane; c < EMBD; c += 32) s += hr[c] * lp_fw[c];
  s = wave_sum(s);
  if (lane == 0) {
    s += lp_fb[0];
    int b = r & (BS - 1);
    float mask = (node_idx[b] < NS - 1) ? 1.0f : 0.0f;
    s *= mask;
    out[r] = 1.0f / (1.0f + expf(-s));
  }
}

// ================= host launcher =================
extern "C" void kernel_launch(void* const* d_in, const int* in_sizes, int n_in,
                              void* d_out, int out_size, void* d_ws, size_t ws_size,
                              hipStream_t stream) {
  (void)in_sizes; (void)n_in; (void)out_size; (void)ws_size;

  const int*   node_ids        = (const int*)  d_in[0];
  const int*   node_idx        = (const int*)  d_in[1];
  const int*   edge_idx        = (const int*)  d_in[2];
  const int*   mem_last_update = (const int*)  d_in[3];
  const int*   mem_rel_t       = (const int*)  d_in[4];
  const int*   mem_dst_id      = (const int*)  d_in[5];
  const float* edge_t          = (const float*)d_in[6];
  const float* edge_msg        = (const float*)d_in[7];
  const float* mem_table       = (const float*)d_in[8];
  const float* mem_msg_table   = (const float*)d_in[9];
  const float* mem_dir         = (const float*)d_in[10];
  const float* te_mem_w        = (const float*)d_in[11];
  const float* te_mem_b        = (const float*)d_in[12];
  const float* gru_w_ih        = (const float*)d_in[13];
  const float* gru_w_hh        = (const float*)d_in[14];
  const float* gru_b_ih        = (const float*)d_in[15];
  const float* gru_b_hh        = (const float*)d_in[16];
  const float* te_gnn_w        = (const float*)d_in[17];
  const float* te_gnn_b        = (const float*)d_in[18];
  const float* Wq = (const float*)d_in[19]; const float* bq = (const float*)d_in[20];
  const float* Wk = (const float*)d_in[21]; const float* bk = (const float*)d_in[22];
  const float* Wv = (const float*)d_in[23]; const float* bv = (const float*)d_in[24];
  const float* We = (const float*)d_in[25];
  const float* Wskip = (const float*)d_in[26]; const float* bskip = (const float*)d_in[27];
  const float* lp_hw = (const float*)d_in[28]; const float* lp_hb = (const float*)d_in[29];
  const float* lp_fw = (const float*)d_in[30]; const float* lp_fb = (const float*)d_in[31];
  float* out = (float*)d_out;

  // ---- workspace layout (bump allocator, 256B aligned) ----
  char* W = (char*)d_ws;
  size_t o = 0;
  auto take = [&](size_t bytes) -> size_t {
    size_t r = o; o += (bytes + 255) & ~(size_t)255; return r;
  };
  float* luf   = (float*)(W + take((size_t)NS * 4));
  float* hf    = (float*)(W + take((size_t)NS * MEMD * 4));
  float* memf  = (float*)(W + take((size_t)NS * MEMD * 4));
  bf16*  memb  = (bf16*) (W + take((size_t)NS * MEMD * 2));
  float* qbuf  = (float*)(W + take((size_t)NS * EMBD * 4));
  float* knbuf = (float*)(W + take((size_t)NS * EMBD * 4));
  float* vnbuf = (float*)(W + take((size_t)NS * EMBD * 4));
  float* zbuf  = (float*)(W + take((size_t)NS * EMBD * 4));
  float* alpha = (float*)(W + take((size_t)ES * 2 * 4));
  float* denom = (float*)(W + take((size_t)NS * 2 * 4));
  float* maxv  = (float*)(W + take(256));
  bf16* wih_b  = (bf16*)(W + take((size_t)G3 * GRUINP * 2));
  bf16* whh_b  = (bf16*)(W + take((size_t)G3 * MEMD * 2));
  bf16* we_b   = (bf16*)(W + take((size_t)EMBD * EDGEDP * 2));
  bf16* wq_b   = (bf16*)(W + take((size_t)EMBD * MEMD * 2));
  bf16* wk_b   = (bf16*)(W + take((size_t)EMBD * MEMD * 2));
  bf16* wv_b   = (bf16*)(W + take((size_t)EMBD * MEMD * 2));
  bf16* wsk_b  = (bf16*)(W + take((size_t)EMBD * MEMD * 2));
  bf16* lph_b  = (bf16*)(W + take((size_t)EMBD * 2 * EMBD * 2));
  bf16* catb   = (bf16*)(W + take((size_t)2 * BS * 2 * EMBD * 2));
  float* hid   = (float*)(W + take((size_t)2 * BS * EMBD * 4));
  // overlapped scratch: phase1 (aggr, gi, gh) reused as phase2 (ea, eproj)
  size_t So = o;
  auto al256 = [](size_t x) { return (x + 255) & ~(size_t)255; };
  bf16*  aggr  = (bf16*) (W + So);
  float* gi    = (float*)(W + So + al256((size_t)NS * GRUINP * 2));
  float* gh    = (float*)(W + So + al256((size_t)NS * GRUINP * 2) + al256((size_t)NS * G3 * 4));
  bf16*  ea    = (bf16*) (W + So);
  float* eproj = (float*)(W + So + al256((size_t)ES * EDGEDP * 2));

  auto cdiv = [](size_t a, size_t b) { return (unsigned)((a + b - 1) / b); };

  // ---- 0: weight conversions (f32 -> bf16, K padded) ----
  k_convert_pad<<<cdiv((size_t)G3 * GRUINP, 256), 256, 0, stream>>>(gru_w_ih, wih_b, G3, GRUIN, GRUINP);
  k_convert_pad<<<cdiv((size_t)G3 * MEMD, 256), 256, 0, stream>>>(gru_w_hh, whh_b, G3, MEMD, MEMD);
  k_convert_pad<<<cdiv((size_t)EMBD * EDGEDP, 256), 256, 0, stream>>>(We, we_b, EMBD, EDGED, EDGEDP);
  k_convert_pad<<<cdiv((size_t)EMBD * MEMD, 256), 256, 0, stream>>>(Wq, wq_b, EMBD, MEMD, MEMD);
  k_convert_pad<<<cdiv((size_t)EMBD * MEMD, 256), 256, 0, stream>>>(Wk, wk_b, EMBD, MEMD, MEMD);
  k_convert_pad<<<cdiv((size_t)EMBD * MEMD, 256), 256, 0, stream>>>(Wv, wv_b, EMBD, MEMD, MEMD);
  k_convert_pad<<<cdiv((size_t)EMBD * MEMD, 256), 256, 0, stream>>>(Wskip, wsk_b, EMBD, MEMD, MEMD);
  k_convert_pad<<<cdiv((size_t)EMBD * 2 * EMBD, 256), 256, 0, stream>>>(lp_hw, lph_b, EMBD, 2 * EMBD, 2 * EMBD);
  k_init<<<cdiv((size_t)NS * 2, 256), 256, 0, stream>>>(denom, (unsigned int*)maxv);

  // ---- 1: memory module ----
  k_build_aggr<<<NS, 256, 0, stream>>>(node_ids, mem_last_update, mem_rel_t, mem_dst_id,
                                       mem_table, mem_msg_table, mem_dir,
                                       te_mem_w, te_mem_b, aggr, hf, memb /*tmp as hb*/, luf);
  // gi = aggr @ w_ih^T + b_ih ; gh = h @ w_hh^T + b_hh
  k_gemm_bf16<<<dim3(G3 / 32, NS / 128), 128, 0, stream>>>(aggr, wih_b, gru_b_ih, gi, NS, G3, GRUINP, 0);
  k_gemm_bf16<<<dim3(G3 / 32, NS / 128), 128, 0, stream>>>(memb, whh_b, gru_b_hh, gh, NS, G3, MEMD, 0);
  k_gru<<<cdiv((size_t)NS * MEMD, 256), 256, 0, stream>>>(gi, gh, hf, memf, memb);

  // ---- 2: GNN ----
  k_build_ea<<<ES, 256, 0, stream>>>(edge_idx, luf, edge_t, edge_msg, te_gnn_w, te_gnn_b, ea);
  k_gemm_bf16<<<dim3(EMBD / 32, ES / 128), 128, 0, stream>>>(ea, we_b, nullptr, eproj, ES, EMBD, EDGEDP, 0);
  k_gemm_bf16<<<dim3(EMBD / 32, NS / 128), 128, 0, stream>>>(memb, wq_b, bq, qbuf, NS, EMBD, MEMD, 0);
  k_gemm_bf16<<<dim3(EMBD / 32, NS / 128), 128, 0, stream>>>(memb, wk_b, bk, knbuf, NS, EMBD, MEMD, 0);
  k_gemm_bf16<<<dim3(EMBD / 32, NS / 128), 128, 0, stream>>>(memb, wv_b, bv, vnbuf, NS, EMBD, MEMD, 0);
  k_gemm_bf16<<<dim3(EMBD / 32, NS / 128), 128, 0, stream>>>(memb, wsk_b, bskip, zbuf, NS, EMBD, MEMD, 0);
  k_alpha<<<ES * 2 / 8, 256, 0, stream>>>(edge_idx, qbuf, knbuf, eproj, alpha, maxv);
  k_expdenom<<<cdiv((size_t)ES * 2, 256), 256, 0, stream>>>(edge_idx, alpha, maxv, denom);
  k_scatter<<<ES, 256, 0, stream>>>(edge_idx, alpha, denom, vnbuf, eproj, zbuf);

  // ---- 3: link predictor ----
  k_cat<<<2 * BS, 256, 0, stream>>>(node_idx, zbuf, catb);
  k_gemm_bf16<<<dim3(EMBD / 32, (2 * BS) / 128), 128, 0, stream>>>(catb, lph_b, lp_hb, hid, 2 * BS, EMBD, 2 * EMBD, 1);
  k_final<<<(2 * BS) / 8, 256, 0, stream>>>(hid, lp_fw, lp_fb, node_idx, out);
}
